// NanoShakes_32366873543378
// MI455X (gfx1250) — matrix operand order repaired and checked by reading, not compiled, not run on hardware
//
#include <hip/hip_runtime.h>
#include <hip/hip_bf16.h>

typedef _Float16 f16;
typedef __attribute__((ext_vector_type(16))) _Float16 v16h;
typedef __attribute__((ext_vector_type(8)))  _Float16 v8h;
typedef __attribute__((ext_vector_type(8)))  float    v8f;

#define WMMA_F32_F16(a, b, c) \
  __builtin_amdgcn_wmma_f32_16x16x32_f16(false, (a), false, (b), (short)0, (c), false, false)

// Async global->LDS copy (CDNA5, tracked by ASYNCcnt). 16B per lane.
#define ASYNC_LOAD_B128(ldsOff, gAddr) \
  asm volatile("global_load_async_to_lds_b128 %0, %1, off" \
               :: "v"(ldsOff), "v"(gAddr) : "memory")
#define WAIT_ASYNCCNT(n) asm volatile("s_wait_asynccnt " #n ::: "memory")

// ---------------------------------------------------------------------------
// WMMA fragment loaders (wave32, 16x16x32 f16).
// A (16x32, MxK): lane l holds M = l%16; K pattern: base=16*(l/16),
//   elems j -> K=base+j (contiguous, matches B).
// B (32x16, KxN): lane l holds N = l%16; elems j -> K = 16*(l/16) + j.
//   Stored as (N,K) row-major so the 16 elems are one contiguous 32B load.
// C/D (16x16 f32): lane l holds N = l%16, rows M = r + 8*(l/16), r=0..7.
// ---------------------------------------------------------------------------
__device__ __forceinline__ v16h load_frag_a(const f16* base, int ld, int r0, int k0) {
  int lane = threadIdx.x & 31;
  int r  = r0 + (lane & 15);
  int kb = k0 + ((lane >> 4) << 4);
  const f16* p = base + (size_t)r * ld + kb;
  v8h lo = *(const v8h*)(p);
  v8h hi = *(const v8h*)(p + 8);
  v16h out;
#pragma unroll
  for (int i = 0; i < 8; ++i) { out[i] = lo[i]; out[i + 8] = hi[i]; }
  return out;
}

__device__ __forceinline__ v16h load_frag_b(const f16* base, int ld, int n0, int k0) {
  int lane = threadIdx.x & 31;
  int n  = n0 + (lane & 15);
  int kb = k0 + ((lane >> 4) << 4);
  return *(const v16h*)(base + (size_t)n * ld + kb);
}

// ---------------------------------------------------------------------------
// fp32 -> f16 weight convert with transpose: Wt[n*K + k] = W[k*N + n]
// ---------------------------------------------------------------------------
__global__ void ns_cvt_t_kernel(const float* __restrict__ W, f16* __restrict__ Wt,
                                int K, int N) {
  size_t g = (size_t)blockIdx.x * 256 + threadIdx.x;
  if (g >= (size_t)K * N) return;
  int    n = (int)(g % N);
  size_t k = g / N;
  Wt[(size_t)n * K + k] = (f16)W[g];
}

// ---------------------------------------------------------------------------
// Embedding: x[b,t,:] = word_emb[tok] + pos_emb[t]
// ---------------------------------------------------------------------------
__global__ void ns_embed_kernel(const int* __restrict__ tokens,
                                const float* __restrict__ we,
                                const float* __restrict__ pe,
                                float* __restrict__ x) {
  const int T = 1024, E = 768;
  int bt = blockIdx.x;
  int t = bt % T;
  int tok = tokens[bt];
  for (int i = threadIdx.x; i < E; i += 256)
    x[(size_t)bt * E + i] = we[(size_t)tok * E + i] + pe[(size_t)t * E + i];
}

// ---------------------------------------------------------------------------
// LayerNorm over E=768, writes f16 (feeds the next WMMA GEMM). 256 thr/row.
// ---------------------------------------------------------------------------
__global__ void ns_layernorm_kernel(const float* __restrict__ x,
                                    const float* __restrict__ g,
                                    const float* __restrict__ bta,
                                    f16* __restrict__ out) {
  const int E = 768;
  int row = blockIdx.x, tid = threadIdx.x;
  const float* xr = x + (size_t)row * E;
  float v0 = xr[tid], v1 = xr[tid + 256], v2 = xr[tid + 512];
  __shared__ float red[256];
  red[tid] = v0 + v1 + v2;
  __syncthreads();
  for (int off = 128; off >= 1; off >>= 1) {
    if (tid < off) red[tid] += red[tid + off];
    __syncthreads();
  }
  float mean = red[0] * (1.0f / E);
  __syncthreads();
  float d0 = v0 - mean, d1 = v1 - mean, d2 = v2 - mean;
  red[tid] = d0 * d0 + d1 * d1 + d2 * d2;
  __syncthreads();
  for (int off = 128; off >= 1; off >>= 1) {
    if (tid < off) red[tid] += red[tid + off];
    __syncthreads();
  }
  float rstd = rsqrtf(red[0] * (1.0f / E) + 1e-5f);
  f16* orow = out + (size_t)row * E;
  orow[tid]       = (f16)(d0 * rstd * g[tid]       + bta[tid]);
  orow[tid + 256] = (f16)(d1 * rstd * g[tid + 256] + bta[tid + 256]);
  orow[tid + 512] = (f16)(d2 * rstd * g[tid + 512] + bta[tid + 512]);
}

// ---------------------------------------------------------------------------
// WMMA GEMM with async double-buffered B staging in LDS.
//   C(MxN) = A(MxK,f16) @ Bt(NxK,f16)^T [+bias][ReLU], out f16 or f32(+resid).
// Block = 8 waves; wave tile 16x64; block tile 64x128; K step 64.
// B tile (128 rows x 64 k = 16KB) staged via global_load_async_to_lds_b128,
// double-buffered with s_wait_asynccnt. A read directly (coalesced b128).
// All 8 B fragments are preloaded before the 8 WMMAs so DS latency overlaps.
// Requires M%64==0, N%128==0, K%64==0.
// ---------------------------------------------------------------------------
__global__ void __launch_bounds__(256)
ns_gemm_wmma_kernel(const f16* __restrict__ A,
                    const f16* __restrict__ Bt,
                    const float* __restrict__ bias,
                    float* __restrict__ Cf,
                    f16* __restrict__ Ch,
                    const float* __restrict__ resid,
                    int M, int N, int K, int relu) {
  __shared__ __attribute__((aligned(128))) f16 btile[2][128 * 64];   // 2 x 16KB

  int tid  = threadIdx.x;
  int lane = tid & 31;
  int w    = tid >> 5;
  int m0   = blockIdx.y * 64 + (w & 3) * 16;
  int nblk = blockIdx.x * 128;
  int ncol = (w >> 2) * 64;

  unsigned ldsBase = (unsigned)(uintptr_t)&btile[0][0];

  // Stage one 128x64 B tile into LDS buffer bufSel (4 async b128 per thread).
  auto stage = [&](int bufSel, int k0) {
    unsigned lb = ldsBase + (unsigned)bufSel * (128u * 64u * 2u);
#pragma unroll
    for (int i = 0; i < 4; ++i) {
      int c   = tid + 256 * i;        // 1024 16B chunks total
      int row = c >> 3;               // 8 chunks per 64-k row
      int kc  = c & 7;
      const f16* g = Bt + (size_t)(nblk + row) * K + k0 + kc * 8;
      unsigned loff = lb + (unsigned)(row * 128 + kc * 16);
      unsigned long long ga = (unsigned long long)(uintptr_t)g;
      ASYNC_LOAD_B128(loff, ga);
    }
  };

  v8f zero = {};
  v8f acc[4] = {zero, zero, zero, zero};

  // One 64-wide K step: preload A frags + all 8 B frags, then 8 WMMAs.
  auto compute = [&](int s) {
    const f16* bt = &btile[s & 1][0];
    int k0 = s << 6;
    v16h a0 = load_frag_a(A, K, m0, k0);
    v16h a1 = load_frag_a(A, K, m0, k0 + 32);
    v16h bf[8];
#pragma unroll
    for (int t = 0; t < 4; ++t) {
      int nl = ncol + 16 * t + (lane & 15);
      const f16* bp = bt + (size_t)nl * 64 + ((lane >> 4) << 4);
      bf[t]     = *(const v16h*)(bp);
      bf[4 + t] = *(const v16h*)(bp + 32);
    }
#pragma unroll
    for (int t = 0; t < 4; ++t) acc[t] = WMMA_F32_F16(a0, bf[t], acc[t]);
#pragma unroll
    for (int t = 0; t < 4; ++t) acc[t] = WMMA_F32_F16(a1, bf[4 + t], acc[t]);
  };

  int nsteps = K >> 6;
  stage(0, 0);
  int s = 0;
  for (; s < nsteps - 1; ++s) {
    stage((s + 1) & 1, (s + 1) << 6);
    WAIT_ASYNCCNT(4);                 // current tile done; next tile in flight
    __syncthreads();
    __builtin_prefetch(A + (size_t)(m0 + (lane & 15)) * K + ((s + 1) << 6), 0, 0);
    compute(s);
    __syncthreads();                  // protect buf before it is re-staged
  }
  WAIT_ASYNCCNT(0);
  __syncthreads();
  compute(s);

  int rowb = m0 + ((lane >> 4) << 3);
#pragma unroll
  for (int t = 0; t < 4; ++t) {
    int n = nblk + ncol + 16 * t + (lane & 15);
    float bv = bias ? bias[n] : 0.0f;
#pragma unroll
    for (int r = 0; r < 8; ++r) {
      float v = acc[t][r] + bv;
      if (relu) v = fmaxf(v, 0.0f);
      size_t idx = (size_t)(rowb + r) * N + n;
      if (Ch) {
        Ch[idx] = (f16)v;
      } else {
        if (resid) v += resid[idx];
        Cf[idx] = v;
      }
    }
  }
}

// ---------------------------------------------------------------------------
// Pack Q/K/V: (B*T,E) f16 -> Q,K packed (B,H,T,64), V transposed (B,H,64,T).
// Q scaled by sqrt(T)=32 (reference multiplies scores by sqrt(T)).
// ---------------------------------------------------------------------------
__global__ void ns_pack_qkv_kernel(const f16* __restrict__ qt,
                                   const f16* __restrict__ kt,
                                   const f16* __restrict__ vt,
                                   f16* __restrict__ qp,
                                   f16* __restrict__ kp,
                                   f16* __restrict__ vp) {
  const int T = 1024, E = 768, H = 12, HS = 64;
  size_t g = (size_t)blockIdx.x * 256 + threadIdx.x;   // over B*T*E
  int e = (int)(g % E);
  size_t bt = g / E;
  int t = (int)(bt % T);
  int b = (int)(bt / T);
  int h = e / HS, d = e % HS;
  size_t bh = (size_t)(b * H + h);
  qp[(bh * T + t) * HS + d] = (f16)((float)qt[g] * 32.0f);
  kp[(bh * T + t) * HS + d] = kt[g];
  vp[(bh * HS + d) * T + t] = vt[g];
}

// ---------------------------------------------------------------------------
// Fused causal attention, flash-style. One wave per (b,h,16-query tile).
// S = Q@K^T via WMMA, online softmax in C-layout registers (16-lane shfl
// reductions), P re-laid out via 16x32 LDS tile into an A-fragment, then
// P@V via WMMA (K = 32 keys/iter). Output f16, concat-heads (B,T,E) layout.
// ---------------------------------------------------------------------------
__global__ void ns_attn_kernel(const f16* __restrict__ Q,
                               const f16* __restrict__ Kc,
                               const f16* __restrict__ Vt,
                               f16* __restrict__ Out) {
  const int T = 1024, HS = 64, H = 12, E = 768;
  int lane = threadIdx.x & 31;
  int qt = blockIdx.x, h = blockIdx.y, b = blockIdx.z;
  int q0 = qt * 16;
  const f16* Qp = Q  + ((size_t)(b * H + h) * T) * HS;
  const f16* Kp = Kc + ((size_t)(b * H + h) * T) * HS;
  const f16* Vp = Vt + ((size_t)(b * H + h) * HS) * T;

  __shared__ __attribute__((aligned(64))) f16 pl[16 * 32];

  v16h qa0 = load_frag_a(Qp, HS, q0, 0);
  v16h qa1 = load_frag_a(Qp, HS, q0, 32);

  v8f zero = {};
  v8f o0 = zero, o1 = zero, o2 = zero, o3 = zero;
  float mrow[8], lrow[8];
#pragma unroll
  for (int r = 0; r < 8; ++r) { mrow[r] = -3.0e38f; lrow[r] = 0.0f; }

  int rowb  = (lane >> 4) << 3;   // 0 or 8
  int colin = lane & 15;
  int kend  = q0 + 16;            // exclusive causal key bound

  for (int j0 = 0; j0 < kend; j0 += 32) {
    bool t1v = (j0 + 16) < kend;
    v8f s0 = zero, s1 = zero;
    {
      v16h kb0 = load_frag_b(Kp, HS, j0, 0);
      v16h kb1 = load_frag_b(Kp, HS, j0, 32);
      s0 = WMMA_F32_F16(qa0, kb0, s0);
      s0 = WMMA_F32_F16(qa1, kb1, s0);
    }
    if (t1v) {
      v16h kb0 = load_frag_b(Kp, HS, j0 + 16, 0);
      v16h kb1 = load_frag_b(Kp, HS, j0 + 16, 32);
      s1 = WMMA_F32_F16(qa0, kb0, s1);
      s1 = WMMA_F32_F16(qa1, kb1, s1);
    }
    int c0 = j0 + colin, c1 = c0 + 16;
#pragma unroll
    for (int r = 0; r < 8; ++r) {
      int rowg = q0 + rowb + r;
      float v0 = (c0 <= rowg) ? s0[r] : -3.0e38f;
      float v1 = (t1v && c1 <= rowg) ? s1[r] : -3.0e38f;
      float mx = fmaxf(v0, v1);
#pragma unroll
      for (int d = 1; d < 16; d <<= 1)
        mx = fmaxf(mx, __shfl_xor(mx, d, 16));
      float mnew  = fmaxf(mrow[r], mx);
      float alpha = __expf(mrow[r] - mnew);
      mrow[r] = mnew;
      float p0 = __expf(v0 - mnew);
      float p1 = t1v ? __expf(v1 - mnew) : 0.0f;
      float sum = p0 + p1;
#pragma unroll
      for (int d = 1; d < 16; d <<= 1)
        sum += __shfl_xor(sum, d, 16);
      lrow[r] = lrow[r] * alpha + sum;
      o0[r] *= alpha; o1[r] *= alpha; o2[r] *= alpha; o3[r] *= alpha;
      pl[(rowb + r) * 32 + colin]      = (f16)p0;
      pl[(rowb + r) * 32 + 16 + colin] = (f16)p1;
    }
    __syncthreads();
    v16h pa  = load_frag_a(pl, 32, 0, 0);
    v16h vb0 = load_frag_b(Vp, T,  0, j0);
    v16h vb1 = load_frag_b(Vp, T, 16, j0);
    v16h vb2 = load_frag_b(Vp, T, 32, j0);
    v16h vb3 = load_frag_b(Vp, T, 48, j0);
    o0 = WMMA_F32_F16(pa, vb0, o0);
    o1 = WMMA_F32_F16(pa, vb1, o1);
    o2 = WMMA_F32_F16(pa, vb2, o2);
    o3 = WMMA_F32_F16(pa, vb3, o3);
    __syncthreads();
  }

  size_t base = ((size_t)b * T + q0 + rowb) * E + h * HS + colin;
#pragma unroll
  for (int r = 0; r < 8; ++r) {
    float inv = 1.0f / lrow[r];
    Out[base + (size_t)r * E + 0]  = (f16)(o0[r] * inv);
    Out[base + (size_t)r * E + 16] = (f16)(o1[r] * inv);
    Out[base + (size_t)r * E + 32] = (f16)(o2[r] * inv);
    Out[base + (size_t)r * E + 48] = (f16)(o3[r] * inv);
  }
}

// ---------------------------------------------------------------------------
// Host launcher
// ---------------------------------------------------------------------------
extern "C" void kernel_launch(void* const* d_in, const int* in_sizes, int n_in,
                              void* d_out, int out_size, void* d_ws, size_t ws_size,
                              hipStream_t stream) {
  (void)in_sizes; (void)n_in; (void)out_size; (void)ws_size;
  const int Vv = 32000, Ee = 768, Ll = 6, Hh = 12, Bb = 4, Tt = 1024;
  const int M = Bb * Tt;               // 4096
  const int E4 = 4 * Ee;               // 3072

  const int*   tokens = (const int*)  d_in[0];
  const float* we     = (const float*)d_in[1];
  const float* pe     = (const float*)d_in[2];
  const float* Wq     = (const float*)d_in[3];
  const float* Wk     = (const float*)d_in[4];
  const float* Wv     = (const float*)d_in[5];
  const float* Wp     = (const float*)d_in[6];
  const float* bproj  = (const float*)d_in[7];
  const float* W1     = (const float*)d_in[8];
  const float* b1     = (const float*)d_in[9];
  const float* W2     = (const float*)d_in[10];
  const float* b2     = (const float*)d_in[11];
  const float* ln1s   = (const float*)d_in[12];
  const float* ln1b   = (const float*)d_in[13];
  const float* ln2s   = (const float*)d_in[14];
  const float* ln2b   = (const float*)d_in[15];
  const float* lnfs   = (const float*)d_in[16];
  const float* lnfb   = (const float*)d_in[17];
  const float* Wout   = (const float*)d_in[18];
  const float* bout   = (const float*)d_in[19];

  char* ws = (char*)d_ws;
  size_t off = 0;
  auto alloc = [&](size_t bytes) -> void* {
    void* p = ws + off;
    off = (off + bytes + 255) & ~(size_t)255;
    return p;
  };

  const size_t e2 = (size_t)Ee * Ee, e4 = (size_t)Ee * E4;
  f16* WqT = (f16*)alloc(Ll * e2 * 2);
  f16* WkT = (f16*)alloc(Ll * e2 * 2);
  f16* WvT = (f16*)alloc(Ll * e2 * 2);
  f16* WpT = (f16*)alloc(Ll * e2 * 2);
  f16* W1T = (f16*)alloc(Ll * e4 * 2);
  f16* W2T = (f16*)alloc(Ll * e4 * 2);
  f16* WoT = (f16*)alloc((size_t)Vv * Ee * 2);
  float* x  = (float*)alloc((size_t)M * Ee * 4);
  f16* hbuf = (f16*)alloc((size_t)M * Ee * 2);
  f16* qtmp = (f16*)alloc((size_t)M * Ee * 2);
  f16* ktmp = (f16*)alloc((size_t)M * Ee * 2);
  f16* vtmp = (f16*)alloc((size_t)M * Ee * 2);
  f16* qp   = (f16*)alloc((size_t)M * Ee * 2);
  f16* kp   = (f16*)alloc((size_t)M * Ee * 2);
  f16* vp   = (f16*)alloc((size_t)M * Ee * 2);
  f16* aout = (f16*)alloc((size_t)M * Ee * 2);
  f16* ffnm = (f16*)alloc((size_t)M * E4 * 2);

  // 1) one-time weight convert+transpose to f16 (N,K)
  for (int l = 0; l < Ll; ++l) {
    ns_cvt_t_kernel<<<(unsigned)((e2 + 255) / 256), 256, 0, stream>>>(Wq + l * e2, WqT + l * e2, Ee, Ee);
    ns_cvt_t_kernel<<<(unsigned)((e2 + 255) / 256), 256, 0, stream>>>(Wk + l * e2, WkT + l * e2, Ee, Ee);
    ns_cvt_t_kernel<<<(unsigned)((e2 + 255) / 256), 256, 0, stream>>>(Wv + l * e2, WvT + l * e2, Ee, Ee);
    ns_cvt_t_kernel<<<(unsigned)((e2 + 255) / 256), 256, 0, stream>>>(Wp + l * e2, WpT + l * e2, Ee, Ee);
    ns_cvt_t_kernel<<<(unsigned)((e4 + 255) / 256), 256, 0, stream>>>(W1 + l * e4, W1T + l * e4, Ee, E4);
    ns_cvt_t_kernel<<<(unsigned)((e4 + 255) / 256), 256, 0, stream>>>(W2 + l * e4, W2T + l * e4, E4, Ee);
  }
  ns_cvt_t_kernel<<<(unsigned)(((size_t)Vv * Ee + 255) / 256), 256, 0, stream>>>(Wout, WoT, Ee, Vv);

  // 2) embedding
  ns_embed_kernel<<<M, 256, 0, stream>>>(tokens, we, pe, x);

  dim3 gE(Ee / 128, M / 64);    // (6, 64)
  dim3 g4E(E4 / 128, M / 64);   // (24, 64)
  dim3 gV(Vv / 128, M / 64);    // (250, 64)
  unsigned packBlocks = (unsigned)((size_t)M * Ee / 256);

  // 3) transformer layers
  for (int l = 0; l < Ll; ++l) {
    ns_layernorm_kernel<<<M, 256, 0, stream>>>(x, ln1s + l * Ee, ln1b + l * Ee, hbuf);
    ns_gemm_wmma_kernel<<<gE, 256, 0, stream>>>(hbuf, WqT + l * e2, nullptr, nullptr, qtmp, nullptr, M, Ee, Ee, 0);
    ns_gemm_wmma_kernel<<<gE, 256, 0, stream>>>(hbuf, WkT + l * e2, nullptr, nullptr, ktmp, nullptr, M, Ee, Ee, 0);
    ns_gemm_wmma_kernel<<<gE, 256, 0, stream>>>(hbuf, WvT + l * e2, nullptr, nullptr, vtmp, nullptr, M, Ee, Ee, 0);
    ns_pack_qkv_kernel<<<packBlocks, 256, 0, stream>>>(qtmp, ktmp, vtmp, qp, kp, vp);
    ns_attn_kernel<<<dim3(Tt / 16, Hh, Bb), 32, 0, stream>>>(qp, kp, vp, aout);
    ns_gemm_wmma_kernel<<<gE, 256, 0, stream>>>(aout, WpT + l * e2, bproj + l * Ee, x, nullptr, x, M, Ee, Ee, 0);
    ns_layernorm_kernel<<<M, 256, 0, stream>>>(x, ln2s + l * Ee, ln2b + l * Ee, hbuf);
    ns_gemm_wmma_kernel<<<g4E, 256, 0, stream>>>(hbuf, W1T + l * e4, b1 + l * E4, nullptr, ffnm, nullptr, M, E4, Ee, 1);
    ns_gemm_wmma_kernel<<<gE, 256, 0, stream>>>(ffnm, W2T + l * e4, b2 + l * Ee, x, nullptr, x, M, Ee, E4, 0);
  }

  // 4) final LN + logits
  ns_layernorm_kernel<<<M, 256, 0, stream>>>(x, lnfs, lnfb, hbuf);
  ns_gemm_wmma_kernel<<<gV, 256, 0, stream>>>(hbuf, WoT, bout, (float*)d_out, nullptr, nullptr, M, Vv, Ee, 0);
}
